// PNN_2319282340353
// MI455X (gfx1250) — matrix operand order repaired and checked by reading, not compiled, not run on hardware
//
#include <hip/hip_runtime.h>
#include <hip/hip_bf16.h>
#include <math.h>

// ---------------------------------------------------------------------------
// PNN forward for MI455X (gfx1250, wave32, WMMA).
// B=4096, F=32, V=50000, E=64, D1=256, DD=16, L2=128.
// Lz/Lp_inner/Lp_outer fused into ONE f16 WMMA GEMM (K=7168) + combined bias;
// W tiles staged to LDS with double-buffered async copies (ASYNCcnt path).
// ---------------------------------------------------------------------------

#define BB   4096
#define FF   32
#define VV   50000
#define EE   64
#define DD1  256
#define DDN  16
#define LL2  128
#define KCAT 7168      // 2048 emb + 1024 inner + 4096 outer
#define KD   288       // 272 (D1+DD) padded to multiple of 32

typedef __attribute__((ext_vector_type(16))) _Float16 v16h;
typedef __attribute__((ext_vector_type(8)))  _Float16 v8h;
typedef __attribute__((ext_vector_type(8)))  float    v8f;
typedef int v4i_vec __attribute__((vector_size(16)));   // matches builtin's int4

union AF { v16h v; v8h h[2]; };

#define AS1 __attribute__((address_space(1)))
#define AS3 __attribute__((address_space(3)))

#if __has_builtin(__builtin_amdgcn_global_load_async_to_lds_b128) && \
    __has_builtin(__builtin_amdgcn_s_wait_asynccnt)
#define USE_ASYNC_LDS 1
#else
#define USE_ASYNC_LDS 0
#endif

// Async 16B global->LDS copy (ASYNCcnt-tracked); sync fallback.
// AS-qualified pointers built via integer round-trip: flat addr == AS1 addr
// for global; flat addr low 32 bits == LDS offset for shared (ISA 10.2).
__device__ __forceinline__ void copy_b128_to_lds(const _Float16* gsrc,
                                                 _Float16* ldst) {
#if USE_ASYNC_LDS
  __builtin_amdgcn_global_load_async_to_lds_b128(
      (AS1 v4i_vec*)(size_t)gsrc, (AS3 v4i_vec*)(size_t)ldst, 0, 0);
#else
  *(v8h*)ldst = *(const v8h*)gsrc;
#endif
}

__device__ __forceinline__ void wait_async_lds() {
#if USE_ASYNC_LDS
  __builtin_amdgcn_s_wait_asynccnt(0);
#endif
}

// ---------------------------------------------------------------------------
// Weight repack: f32 -> f16; concat [Wz|Wpi|Wpo] rows; transpose W1/W2 to
// (N x K) row-major (so WMMA B-fragments are contiguous loads); combined bias.
// ---------------------------------------------------------------------------
__global__ __launch_bounds__(256) void prep_weights(
    const float* __restrict__ Wz, const float* __restrict__ Wpi,
    const float* __restrict__ Wpo, const float* __restrict__ bz,
    const float* __restrict__ bpi, const float* __restrict__ bpo,
    const float* __restrict__ B1, const float* __restrict__ W1,
    const float* __restrict__ W2,
    _Float16* __restrict__ Wcat, _Float16* __restrict__ W1T,
    _Float16* __restrict__ W2T, float* __restrict__ bias_cat) {
  const int NW  = DD1 * KCAT;          // 1,835,008
  const int NW1 = DD1 * KD;            // 73,728
  const int NW2 = LL2 * DD1;           // 32,768
  int i = blockIdx.x * blockDim.x + threadIdx.x;
  if (i < NW) {
    int d = i / KCAT, c = i % KCAT;
    float v;
    if (c < 2048)      v = Wz [(size_t)d * 2048 + c];
    else if (c < 3072) v = Wpi[(size_t)d * 1024 + (c - 2048)];
    else               v = Wpo[(size_t)d * 4096 + (c - 3072)];
    Wcat[i] = (_Float16)v;
  } else if (i < NW + NW1) {
    int j = i - NW; int nn = j / KD, k = j % KD;         // W1 is (272,256) K-major
    W1T[j] = (k < 272) ? (_Float16)W1[(size_t)k * DD1 + nn] : (_Float16)0.0f;
  } else if (i < NW + NW1 + NW2) {
    int j = i - NW - NW1; int nn = j / DD1, k = j % DD1; // W2 is (256,128)
    W2T[j] = (_Float16)W2[(size_t)k * LL2 + nn];
  } else if (i < NW + NW1 + NW2 + DD1) {
    int d = i - (NW + NW1 + NW2);
    bias_cat[d] = bz[d] + bpi[d] + bpo[d] + B1[d];
  }
}

// ---------------------------------------------------------------------------
// Per-sample feature build: gather embeddings (f32 tables -> f16), field-sum s,
// inner = emb*embT via WMMA from LDS, outer = s (x) s. One block per sample.
// Waves 0-3 run WMMA (EXEC all-ones per wave); waves 4-7 emit the outer tile.
// ---------------------------------------------------------------------------
__global__ __launch_bounds__(256) void build_features(
    const int* __restrict__ sparse, const float* __restrict__ tables,
    const float* __restrict__ dense,
    _Float16* __restrict__ X, _Float16* __restrict__ deepx) {
  __shared__ _Float16 Eh[FF][EE];   // 4 KB, row stride 128B keeps 32B alignment
  __shared__ float    sld[EE];

  const int b = blockIdx.x;
  const int t = threadIdx.x;

  // ---- gather: thread t -> field t>>3, 8 consecutive elements -------------
  {
    int f = t >> 3, c = (t & 7) * 8;
    int idx = sparse[b * FF + f];
    const float* row = tables + ((size_t)f * VV + idx) * EE + c;
    float4 r0 = *(const float4*)(row);
    float4 r1 = *(const float4*)(row + 4);
    v8h hv;
    hv[0] = (_Float16)r0.x; hv[1] = (_Float16)r0.y;
    hv[2] = (_Float16)r0.z; hv[3] = (_Float16)r0.w;
    hv[4] = (_Float16)r1.x; hv[5] = (_Float16)r1.y;
    hv[6] = (_Float16)r1.z; hv[7] = (_Float16)r1.w;
    *(v8h*)&Eh[f][c] = hv;                                     // ds_store_b128
    *(v8h*)(X + (size_t)b * KCAT + f * EE + c) = hv;           // emb segment
  }
  // dense tail of deep_x (cols 256..271) + zero pad (272..287); independent
  if (t >= 224 && t < 240)
    deepx[(size_t)b * KD + 256 + (t - 224)] = (_Float16)dense[b * DDN + (t - 224)];
  else if (t >= 240)
    deepx[(size_t)b * KD + 272 + (t - 240)] = (_Float16)0.0f;

  __syncthreads();

  // ---- s[e] = sum over fields -------------------------------------------
  if (t < EE) {
    float s = 0.f;
    #pragma unroll
    for (int f = 0; f < FF; ++f) s += (float)Eh[f][t];
    sld[t] = s;
  }
  __syncthreads();

  const int wave = t >> 5, lane = t & 31;
  if (wave < 4) {
    // ---- inner = Eh (32x64) @ EhT (64x32): 4 output tiles, K in 2 steps ---
    const int i0 = (wave & 1) * 16, j0 = (wave >> 1) * 16;
    const int m = lane & 15, kh = lane >> 4;
    const int n = lane & 15, kb = (lane >> 4) * 16;
    v8f acc = {};
    #pragma unroll
    for (int k0 = 0; k0 < EE; k0 += 32) {
      AF a;
      a.h[0] = *(const v8h*)&Eh[i0 + m][k0 + kh * 16];         // ds_load_b128
      a.h[1] = *(const v8h*)&Eh[i0 + m][k0 + kh * 16 + 8];
      v16h bf = *(const v16h*)&Eh[j0 + n][k0 + kb];            // row n = col n of EhT
      acc = __builtin_amdgcn_wmma_f32_16x16x32_f16(
          false, a.v, false, bf, (short)0, acc, false, false);
    }
    const int mh = lane >> 4;
    _Float16* ip = X + (size_t)b * KCAT + 2048;                // inner segment
    #pragma unroll
    for (int v = 0; v < 8; ++v)
      ip[(i0 + mh * 8 + v) * FF + (j0 + n)] = (_Float16)acc[v];
  } else {
    // ---- outer = s (x) s : 4096 f16 elements, 128 threads x 32 -----------
    const int t2 = t - 128;
    _Float16* op = X + (size_t)b * KCAT + 3072;                // outer segment
    #pragma unroll
    for (int q = 0; q < 32; ++q) {
      int id = t2 + q * 128;
      op[id] = (_Float16)(sld[id >> 6] * sld[id & 63]);
    }
  }
}

// ---------------------------------------------------------------------------
// Generic f16 WMMA GEMM: C[m,n] = sum_k X[m,k]*W[n,k] + bias[n], optional tanh.
// X: (M x K) row-major f16, streamed from global per wave (16B chunks/lane).
// W: (N x K) row-major f16, staged per-block into LDS as double-buffered
// 64x32 tiles using async global->LDS copies (ASYNCcnt) when available.
// Block = 256 threads = 8 waves; tile = 128(M) x 64(N); wave -> 16x64 strip.
// ---------------------------------------------------------------------------
template <bool TANH>
__global__ __launch_bounds__(256) void gemm_wmma(
    const _Float16* __restrict__ X, const _Float16* __restrict__ W,
    const float* __restrict__ bias, _Float16* __restrict__ C,
    int ldc, int K) {
  __shared__ _Float16 Wlds[2][64][32];   // 2-stage 64(N) x 32(K) tile, 8 KB

  const int wave = threadIdx.x >> 5;
  const int lane = threadIdx.x & 31;
  const int r0 = blockIdx.x * 128 + wave * 16;
  const int c0 = blockIdx.y * 64;

  const int m  = lane & 15, kh = lane >> 4;        // A-fragment mapping
  const int n  = lane & 15, kb = (lane >> 4) * 16; // B-fragment mapping

  // cooperative W-tile staging: thread -> (row sn, 16B chunk sq)
  const int sn = threadIdx.x >> 2, sq = (threadIdx.x & 3) * 8;
  const _Float16* wrow = W + (size_t)(c0 + sn) * K + sq;

  v8f acc[4] = {{}, {}, {}, {}};
  const _Float16* ap = X + (size_t)(r0 + m) * K + kh * 16;

  // prologue: stage K-chunk 0 into buffer 0 (one async op per wave)
  copy_b128_to_lds(wrow, &Wlds[0][sn][sq]);

  for (int k0 = 0; k0 < K; k0 += 32) {
    const int cur = (k0 >> 5) & 1;
    wait_async_lds();                      // this wave's slice of `cur` landed
    __syncthreads();                       // tile ready; prior reads of ^1 done
    if (k0 + 32 < K)                       // stage next chunk into other buffer
      copy_b128_to_lds(wrow + (k0 + 32), &Wlds[cur ^ 1][sn][sq]);

    AF a;
    a.h[0] = *(const v8h*)(ap);            // K = kh*16 .. +7
    a.h[1] = *(const v8h*)(ap + 8);        // K = kh*16+8 .. +7
    __builtin_prefetch(ap + 256, 0, 0);    // global_prefetch_b8 (stream X)
    #pragma unroll
    for (int tt = 0; tt < 4; ++tt) {
      const v16h bf = *(const v16h*)&Wlds[cur][tt * 16 + n][kb];
      acc[tt] = __builtin_amdgcn_wmma_f32_16x16x32_f16(
          false, a.v, false, bf, (short)0, acc[tt], false, false);
    }
    ap += 32;
  }

  const int mh = lane >> 4;
  #pragma unroll
  for (int tt = 0; tt < 4; ++tt) {
    const int col = c0 + tt * 16 + n;
    const float bv = bias[col];
    #pragma unroll
    for (int v = 0; v < 8; ++v) {
      float x = acc[tt][v] + bv;
      if (TANH) x = tanhf(x);
      C[(size_t)(r0 + mh * 8 + v) * ldc + col] = (_Float16)x;
    }
  }
}

// ---------------------------------------------------------------------------
// Final: out[b] = dot(h2[b,:], W3) + b3. One 32-lane group per row.
// ---------------------------------------------------------------------------
__global__ __launch_bounds__(256) void final_out(
    const _Float16* __restrict__ h2, const float* __restrict__ W3,
    const float* __restrict__ b3, float* __restrict__ out) {
  const int t = threadIdx.x;
  const int w = t >> 5, lane = t & 31;
  const int b = blockIdx.x * 8 + w;
  const _Float16* hp = h2 + (size_t)b * LL2;
  float acc = 0.f;
  #pragma unroll
  for (int q = 0; q < 4; ++q) {
    int e = lane + q * 32;
    acc += (float)hp[e] * W3[e];
  }
  #pragma unroll
  for (int off = 16; off > 0; off >>= 1)
    acc += __shfl_down(acc, off, 32);
  if (lane == 0) out[b] = acc + b3[0];
}

// ---------------------------------------------------------------------------
extern "C" void kernel_launch(void* const* d_in, const int* in_sizes, int n_in,
                              void* d_out, int out_size, void* d_ws, size_t ws_size,
                              hipStream_t stream) {
  const int*   sparse = (const int*)  d_in[0];
  const float* dense  = (const float*)d_in[1];
  const float* tables = (const float*)d_in[2];
  const float* Wz     = (const float*)d_in[3];
  const float* bz     = (const float*)d_in[4];
  const float* Wpi    = (const float*)d_in[5];
  const float* bpi    = (const float*)d_in[6];
  const float* Wpo    = (const float*)d_in[7];
  const float* bpo    = (const float*)d_in[8];
  const float* B1     = (const float*)d_in[9];
  const float* W1     = (const float*)d_in[10];
  const float* b1     = (const float*)d_in[11];
  const float* W2     = (const float*)d_in[12];
  const float* b2     = (const float*)d_in[13];
  const float* W3     = (const float*)d_in[14];
  const float* b3     = (const float*)d_in[15];
  float* out = (float*)d_out;

  char* ws = (char*)d_ws;
  size_t off = 0;
  _Float16* X     = (_Float16*)(ws + off); off += (size_t)BB * KCAT * 2;  // 58.7 MB
  _Float16* Wcat  = (_Float16*)(ws + off); off += (size_t)DD1 * KCAT * 2; // 3.7 MB
  _Float16* deepx = (_Float16*)(ws + off); off += (size_t)BB * KD * 2;    // 2.4 MB
  _Float16* h1    = (_Float16*)(ws + off); off += (size_t)BB * DD1 * 2;   // 2.1 MB
  _Float16* h2    = (_Float16*)(ws + off); off += (size_t)BB * LL2 * 2;   // 1.0 MB
  _Float16* W1T   = (_Float16*)(ws + off); off += (size_t)DD1 * KD * 2;
  _Float16* W2T   = (_Float16*)(ws + off); off += (size_t)LL2 * DD1 * 2;
  float*    bcat  = (float*)   (ws + off); off += (size_t)DD1 * 4;

  const int prep_n = DD1 * KCAT + DD1 * KD + LL2 * DD1 + DD1;
  prep_weights<<<(prep_n + 255) / 256, 256, 0, stream>>>(
      Wz, Wpi, Wpo, bz, bpi, bpo, B1, W1, W2, Wcat, W1T, W2T, bcat);

  build_features<<<BB, 256, 0, stream>>>(sparse, tables, dense, X, deepx);

  // L1 = X @ WcatT + (bz+bpi+bpo+B1), written into deep_x cols [0,256)
  gemm_wmma<false><<<dim3(BB / 128, DD1 / 64), 256, 0, stream>>>(
      X, Wcat, bcat, deepx, KD, KCAT);

  // h1 = tanh(deep_x @ W1 + b1)
  gemm_wmma<true><<<dim3(BB / 128, DD1 / 64), 256, 0, stream>>>(
      deepx, W1T, b1, h1, DD1, KD);

  // h2 = tanh(h1 @ W2 + b2)
  gemm_wmma<true><<<dim3(BB / 128, LL2 / 64), 256, 0, stream>>>(
      h1, W2T, b2, h2, LL2, DD1);

  final_out<<<BB / 8, 256, 0, stream>>>(h2, W3, b3, out);
}
